// BigBirdProteinModel_33612414059115
// MI455X (gfx1250) — matrix operand order repaired and checked by reading, not compiled, not run on hardware
//
#include <hip/hip_runtime.h>
#include <hip/hip_bf16.h>

// ---------------------------------------------------------------------------
// BigBird protein model forward for MI455X (gfx1250), wave32 + WMMA bf16.
// GEMMs: double-buffered LDS, async global->LDS (ASYNCcnt) when available.
// ---------------------------------------------------------------------------

#define S_LEN   4096
#define IN_DIM  1024
#define D_MODEL 768
#define N_HEAD  12
#define DH      64
#define BS_BLK  64
#define NB_BLK  64
#define FF_DIM  3072
#define N_LAYER 12
#define H1_DIM  512
#define TGT_DIM 128
#define R_RAND  3
#define BL_STRIDE 72     // ints per (head, qblock) entry in block list

#define GEMM_KSTEP  32
#define GEMM_STRIDE 40   // 32 + 8 pad: keeps 16B alignment, avoids bank conflicts

typedef __attribute__((ext_vector_type(16))) __bf16 v16bf;
typedef __attribute__((ext_vector_type(8)))  float  v8f;
typedef __attribute__((ext_vector_type(4)))  int    vi4;

struct FragBits { uint4 lo, hi; };

#if defined(__HIP_DEVICE_COMPILE__) && defined(__gfx1250__) && \
    __has_builtin(__builtin_amdgcn_global_load_async_to_lds_b128)
#define HAVE_ASYNC_LDS 1
#else
#define HAVE_ASYNC_LDS 0
#endif

#if HAVE_ASYNC_LDS
__device__ __forceinline__ void async_ld_b128(const void* g, void* l) {
    typedef __attribute__((address_space(1))) vi4 gvi4;   // global
    typedef __attribute__((address_space(3))) vi4 lvi4;   // LDS
    __builtin_amdgcn_global_load_async_to_lds_b128(
        (gvi4*)g, (lvi4*)l, /*offset=*/0, /*cpol=*/0);
}
#endif
__device__ __forceinline__ void wait_async_hi() {
#if HAVE_ASYNC_LDS
    asm volatile("s_wait_asynccnt 0x8" ::: "memory");
#endif
}
__device__ __forceinline__ void wait_async_zero() {
#if HAVE_ASYNC_LDS
    asm volatile("s_wait_asynccnt 0x0" ::: "memory");
#endif
}

__device__ __forceinline__ unsigned short f2bf(float f) {
    unsigned int u = __float_as_uint(f);
    unsigned int r = u + 0x7FFFu + ((u >> 16) & 1u);   // round to nearest even
    return (unsigned short)(r >> 16);
}
__device__ __forceinline__ float gelu_tanh(float x) {
    float x3 = x * x * x;
    return 0.5f * x * (1.0f + tanhf(0.7978845608028654f * (x + 0.044715f * x3)));
}

__device__ __forceinline__ v8f wmma_bf16(v16bf a, v16bf b, v8f c) {
    return __builtin_amdgcn_wmma_f32_16x16x32_bf16(
        false, a, false, b, (short)0, c, false, false);
}

// A fragment (16x32, M=lane&15): elems 0..7 -> K = k0 + half*8, 8..15 -> +16.
__device__ __forceinline__ v16bf load_frag_a(const unsigned short* base, int stride,
                                             int lane, int k0) {
    int row = lane & 15, half = lane >> 4;
    const unsigned short* p = base + row * stride + k0 + half * 8;
    FragBits f;
    f.lo = *reinterpret_cast<const uint4*>(p);
    f.hi = *reinterpret_cast<const uint4*>(p + 16);
    return __builtin_bit_cast(v16bf, f);
}
// B fragment from Bt[N][K]: N=lane&15, elems 0..7 -> K = k0 + half*8, 8..15 -> +16.
__device__ __forceinline__ v16bf load_frag_b(const unsigned short* base, int stride,
                                             int lane, int k0) {
    int col = lane & 15, half = lane >> 4;
    const unsigned short* p = base + col * stride + k0 + half * 8;
    FragBits f;
    f.lo = *reinterpret_cast<const uint4*>(p);
    f.hi = *reinterpret_cast<const uint4*>(p + 16);
    return __builtin_bit_cast(v16bf, f);
}

// ---------------------------------------------------------------------------
// Generic bf16 GEMM: C[M,N] = A[M,K] * Bt[N,K]^T + bias, optional activation.
// 128 threads (4 waves); 128x128 tile, double-buffered 32-wide K slabs.
// act: 0 = none, 1 = gelu(tanh), 2 = relu.
// ---------------------------------------------------------------------------
__global__ __launch_bounds__(128)
void gemm_bf16(const unsigned short* __restrict__ Ag,
               const unsigned short* __restrict__ Bt,
               const float* __restrict__ bias,
               float* __restrict__ Cf,
               unsigned short* __restrict__ Cb,
               int M, int N, int K, int act) {
    __shared__ __align__(16) unsigned short As[2][128 * GEMM_STRIDE];
    __shared__ __align__(16) unsigned short Bs[2][128 * GEMM_STRIDE];

    const int tid  = threadIdx.x;
    const int lane = tid & 31;
    const int wave = tid >> 5;
    const int m0 = blockIdx.y * 128;
    const int n0 = blockIdx.x * 128;
    const int wm = (wave >> 1) * 64;
    const int wn = (wave & 1) * 64;

    // Stage one 128x32 slab of A and Bt into buffer `bufi`.
    auto stage = [&](int k0, int bufi) {
#pragma unroll
        for (int i = 0; i < 4; ++i) {
            int idx = tid + i * 128;          // 0..511
            int r = idx >> 2, c = (idx & 3) * 8;
            int ra = m0 + r; if (ra > M - 1) ra = M - 1;   // clamp: masked later
            int rb = n0 + r; if (rb > N - 1) rb = N - 1;
#if HAVE_ASYNC_LDS
            async_ld_b128(Ag + (size_t)ra * K + k0 + c, &As[bufi][r * GEMM_STRIDE + c]);
            async_ld_b128(Bt + (size_t)rb * K + k0 + c, &Bs[bufi][r * GEMM_STRIDE + c]);
#else
            *reinterpret_cast<uint4*>(&As[bufi][r * GEMM_STRIDE + c]) =
                *reinterpret_cast<const uint4*>(Ag + (size_t)ra * K + k0 + c);
            *reinterpret_cast<uint4*>(&Bs[bufi][r * GEMM_STRIDE + c]) =
                *reinterpret_cast<const uint4*>(Bt + (size_t)rb * K + k0 + c);
            if (k0 + GEMM_KSTEP < K)
                __builtin_prefetch(Ag + (size_t)ra * K + k0 + GEMM_KSTEP, 0, 0);
#endif
        }
    };

    v8f acc[4][4] = {};
    const int T = K / GEMM_KSTEP;

    stage(0, 0);
    for (int t = 0; t < T; ++t) {
        if (t + 1 < T) {
            stage((t + 1) * GEMM_KSTEP, (t + 1) & 1);
            wait_async_hi();      // own 8 loads of slab t complete; t+1 in flight
        } else {
            wait_async_zero();
        }
        __syncthreads();          // all waves' slab-t DMA/stores visible

        const unsigned short* Ab = &As[t & 1][0];
        const unsigned short* Bb = &Bs[t & 1][0];
        v16bf af[4], bf[4];
#pragma unroll
        for (int i = 0; i < 4; ++i)
            af[i] = load_frag_a(Ab + (wm + i * 16) * GEMM_STRIDE, GEMM_STRIDE, lane, 0);
#pragma unroll
        for (int j = 0; j < 4; ++j)
            bf[j] = load_frag_b(Bb + (wn + j * 16) * GEMM_STRIDE, GEMM_STRIDE, lane, 0);
#pragma unroll
        for (int i = 0; i < 4; ++i)
#pragma unroll
            for (int j = 0; j < 4; ++j)
                acc[i][j] = wmma_bf16(af[i], bf[j], acc[i][j]);
        __syncthreads();          // done reading buf before it is re-staged
    }

    // Epilogue. C layout: lane = column, VGPR r = row (m = r + 8*(lane>>4)).
    const int cl = lane & 15, ch = lane >> 4;
#pragma unroll
    for (int i = 0; i < 4; ++i) {
#pragma unroll
        for (int j = 0; j < 4; ++j) {
#pragma unroll
            for (int r = 0; r < 8; ++r) {
                int row = m0 + wm + i * 16 + r + ch * 8;
                int col = n0 + wn + j * 16 + cl;
                if (row < M && col < N) {
                    float v = acc[i][j][r];
                    if (bias) v += bias[col];
                    if (act == 1) v = gelu_tanh(v);
                    else if (act == 2) v = fmaxf(v, 0.0f);
                    if (Cf) Cf[(size_t)row * N + col] = v;
                    if (Cb) Cb[(size_t)row * N + col] = f2bf(v);
                }
            }
        }
    }
}

// ---------------------------------------------------------------------------
// Block-sparse attention (flash-style online softmax) for one (head, qblock).
// 128 threads = 4 waves; wave w owns query rows [w*16, w*16+16).
// Q,K,V are bf16 [S, D] with head h in columns h*64..h*64+63.
// ---------------------------------------------------------------------------
__global__ __launch_bounds__(128)
void bigbird_attn(const unsigned short* __restrict__ Q,
                  const unsigned short* __restrict__ Kg,
                  const unsigned short* __restrict__ Vg,
                  const float* __restrict__ mask_add,
                  const int* __restrict__ blist,
                  unsigned short* __restrict__ ctx) {
    const int head = blockIdx.x;
    const int qb   = blockIdx.y;

    __shared__ __align__(16) unsigned short Qs[64 * 72];
    __shared__ __align__(16) unsigned short Ks[64 * 72];
    __shared__ __align__(16) unsigned short Vt[64 * 72];   // transposed: [dh][key]
    __shared__ __align__(16) unsigned short Ps[4][16 * 72];

    const int tid  = threadIdx.x;
    const int lane = tid & 31;
    const int wave = tid >> 5;
    const int cl = lane & 15, ch = lane >> 4;
    const float scale = 0.125f;  // 1/sqrt(64)

    const int* ent = blist + (head * NB_BLK + qb) * BL_STRIDE;
    const int cnt = ent[0];

    // Stage Q block (64 rows x 64 dh).
#pragma unroll
    for (int i = 0; i < 4; ++i) {
        int idx = tid + i * 128;            // 0..511
        int r = idx >> 3, c = (idx & 7) * 8;
        uint4 v = *reinterpret_cast<const uint4*>(
            Q + (size_t)(qb * BS_BLK + r) * D_MODEL + head * DH + c);
        *reinterpret_cast<uint4*>(Qs + r * 72 + c) = v;
    }

    v8f o[4] = {};                // O slice: rows 16 x dh 64
    float m_run[8], l_run[8];
#pragma unroll
    for (int r = 0; r < 8; ++r) { m_run[r] = -1e30f; l_run[r] = 0.0f; }

    for (int it = 0; it < cnt; ++it) {
        const int kb = ent[1 + it];
        // Stage K block (as Bt for Q*K^T) and V block transposed.
#pragma unroll
        for (int i = 0; i < 4; ++i) {
            int idx = tid + i * 128;
            int r = idx >> 3, c = (idx & 7) * 8;
            uint4 kv = *reinterpret_cast<const uint4*>(
                Kg + (size_t)(kb * BS_BLK + r) * D_MODEL + head * DH + c);
            *reinterpret_cast<uint4*>(Ks + r * 72 + c) = kv;
            uint4 vv = *reinterpret_cast<const uint4*>(
                Vg + (size_t)(kb * BS_BLK + r) * D_MODEL + head * DH + c);
            const unsigned short* ve = reinterpret_cast<const unsigned short*>(&vv);
#pragma unroll
            for (int e = 0; e < 8; ++e)
                Vt[(c + e) * 72 + r] = ve[e];      // Vt[dh][key]
        }
        __syncthreads();

        // ---- S = scale * Q Kt + mask : 16x64 slice per wave ----
        v8f st[4] = {};
#pragma unroll
        for (int kk = 0; kk < 64; kk += 32) {
            v16bf aq = load_frag_a(Qs + (wave * 16) * 72, 72, lane, kk);
#pragma unroll
            for (int j = 0; j < 4; ++j) {
                v16bf bk = load_frag_b(Ks + (j * 16) * 72, 72, lane, kk);
                st[j] = wmma_bf16(aq, bk, st[j]);
            }
        }
        float mv[4];
#pragma unroll
        for (int j = 0; j < 4; ++j)
            mv[j] = mask_add[kb * BS_BLK + j * 16 + cl];
#pragma unroll
        for (int j = 0; j < 4; ++j)
#pragma unroll
            for (int r = 0; r < 8; ++r)
                st[j][r] = st[j][r] * scale + mv[j];

        // ---- online softmax (row = r + 8*ch, col = j*16 + cl) ----
        float alpha[8];
#pragma unroll
        for (int r = 0; r < 8; ++r) {
            float mx = fmaxf(fmaxf(st[0][r], st[1][r]), fmaxf(st[2][r], st[3][r]));
#pragma unroll
            for (int off = 8; off >= 1; off >>= 1)
                mx = fmaxf(mx, __shfl_xor(mx, off, 32));
            float mn = fmaxf(m_run[r], mx);
            alpha[r] = __expf(m_run[r] - mn);
            m_run[r] = mn;
#pragma unroll
            for (int j = 0; j < 4; ++j)
                st[j][r] = __expf(st[j][r] - mn);
            float rs = st[0][r] + st[1][r] + st[2][r] + st[3][r];
#pragma unroll
            for (int off = 8; off >= 1; off >>= 1)
                rs += __shfl_xor(rs, off, 32);
            l_run[r] = l_run[r] * alpha[r] + rs;
        }
        // Rescale O, dump P (bf16) to LDS in row-major [16][72].
#pragma unroll
        for (int j = 0; j < 4; ++j)
#pragma unroll
            for (int r = 0; r < 8; ++r) {
                o[j][r] *= alpha[r];
                Ps[wave][(r + ch * 8) * 72 + j * 16 + cl] = f2bf(st[j][r]);
            }

        // ---- O += P * V ----
#pragma unroll
        for (int kk = 0; kk < 64; kk += 32) {
            v16bf ap = load_frag_a(&Ps[wave][0], 72, lane, kk);
#pragma unroll
            for (int j = 0; j < 4; ++j) {
                v16bf bv = load_frag_b(Vt + (j * 16) * 72, 72, lane, kk);
                o[j] = wmma_bf16(ap, bv, o[j]);
            }
        }
        __syncthreads();   // everyone done with Ks/Vt before next overwrite
    }

    // Final normalize and store ctx (bf16, head-major layout [s][head*64+dh]).
    float inv_l[8];
#pragma unroll
    for (int r = 0; r < 8; ++r) inv_l[r] = 1.0f / l_run[r];
#pragma unroll
    for (int j = 0; j < 4; ++j)
#pragma unroll
        for (int r = 0; r < 8; ++r) {
            int row = qb * BS_BLK + wave * 16 + r + ch * 8;
            int col = head * DH + j * 16 + cl;
            ctx[(size_t)row * D_MODEL + col] = f2bf(o[j][r] * inv_l[r]);
        }
}

// ---------------------------------------------------------------------------
// Small elementwise / utility kernels
// ---------------------------------------------------------------------------
__global__ void mask_kernel(const int* __restrict__ am, float* __restrict__ madd, int n) {
    int t = blockIdx.x * blockDim.x + threadIdx.x;
    if (t < n) madd[t] = (1.0f - (float)am[t]) * -1e9f;
}
__global__ void f32_to_bf16_kernel(const float* __restrict__ x,
                                   unsigned short* __restrict__ y, int n) {
    int t = blockIdx.x * blockDim.x + threadIdx.x;
    if (t < n) y[t] = f2bf(x[t]);
}
// W[K,N] fp32 -> Wt[N,K] bf16
__global__ void wt_kernel(const float* __restrict__ W, unsigned short* __restrict__ Wt,
                          int K, int N) {
    int t = blockIdx.x * blockDim.x + threadIdx.x;
    if (t < K * N) {
        int k = t / N, n = t % N;
        Wt[(size_t)n * K + k] = f2bf(W[t]);
    }
}
__global__ void embadd_kernel(float* __restrict__ h, const float* __restrict__ pos,
                              const float* __restrict__ typ, int n, int dcols) {
    int t = blockIdx.x * blockDim.x + threadIdx.x;
    if (t < n) h[t] += pos[t] + typ[t % dcols];
}
__global__ void build_blist(const int* __restrict__ rb, int* __restrict__ blist) {
    int t = blockIdx.x * blockDim.x + threadIdx.x;
    if (t >= N_HEAD * NB_BLK) return;
    int h = t / NB_BLK, m = t % NB_BLK;
    int* e = blist + t * BL_STRIDE;
    if (m == 0 || m == NB_BLK - 1) {
        e[0] = NB_BLK;
        for (int i = 0; i < NB_BLK; ++i) e[1 + i] = i;
    } else {
        e[0] = 8;
        e[1] = m - 1; e[2] = m; e[3] = m + 1; e[4] = 0; e[5] = NB_BLK - 1;
        const int* r = rb + ((size_t)h * (NB_BLK - 2) + (m - 1)) * R_RAND;
        e[6] = r[0]; e[7] = r[1]; e[8] = r[2];
    }
}
// LayerNorm (optionally fused residual): out = LN(x + resid) * g + b
__global__ __launch_bounds__(256)
void ln_kernel(const float* __restrict__ x, const float* __restrict__ resid,
               const float* __restrict__ g, const float* __restrict__ b,
               float* __restrict__ out, unsigned short* __restrict__ obf, int C) {
    int row = blockIdx.x;
    const float* xr = x + (size_t)row * C;
    const float* rr = resid ? resid + (size_t)row * C : nullptr;
    float s = 0.f, s2 = 0.f;
    for (int c = threadIdx.x; c < C; c += blockDim.x) {
        float v = xr[c] + (rr ? rr[c] : 0.f);
        s += v; s2 += v * v;
    }
#pragma unroll
    for (int off = 16; off >= 1; off >>= 1) {
        s += __shfl_xor(s, off, 32);
        s2 += __shfl_xor(s2, off, 32);
    }
    __shared__ float rs[8], rs2[8];
    int w = threadIdx.x >> 5;
    if ((threadIdx.x & 31) == 0) { rs[w] = s; rs2[w] = s2; }
    __syncthreads();
    if (threadIdx.x == 0) {
        float ts = 0.f, t2 = 0.f;
        for (int i = 0; i < 8; ++i) { ts += rs[i]; t2 += rs2[i]; }
        rs[0] = ts; rs2[0] = t2;
    }
    __syncthreads();
    float mu = rs[0] / C;
    float var = rs2[0] / C - mu * mu;
    float inv = rsqrtf(var + 1e-12f);
    for (int c = threadIdx.x; c < C; c += blockDim.x) {
        float v = xr[c] + (rr ? rr[c] : 0.f);
        float y = (v - mu) * inv * g[c] + b[c];
        out[(size_t)row * C + c] = y;
        if (obf) obf[(size_t)row * C + c] = f2bf(y);
    }
}

// ---------------------------------------------------------------------------
// Host-side orchestration
// ---------------------------------------------------------------------------
extern "C" void kernel_launch(void* const* d_in, const int* in_sizes, int n_in,
                              void* d_out, int out_size, void* d_ws, size_t ws_size,
                              hipStream_t stream) {
    (void)in_sizes; (void)n_in; (void)out_size; (void)ws_size;

    const float* x        = (const float*)d_in[0];
    const int*   am       = (const int*)d_in[1];
    const float* proj_w   = (const float*)d_in[2];
    const float* proj_b   = (const float*)d_in[3];
    const float* pos_emb  = (const float*)d_in[4];
    const float* type_emb = (const float*)d_in[5];
    const float* emb_g    = (const float*)d_in[6];
    const float* emb_b    = (const float*)d_in[7];
    const float* wq = (const float*)d_in[8];   const float* bq = (const float*)d_in[9];
    const float* wk = (const float*)d_in[10];  const float* bk = (const float*)d_in[11];
    const float* wv = (const float*)d_in[12];  const float* bv = (const float*)d_in[13];
    const float* wo = (const float*)d_in[14];  const float* bo = (const float*)d_in[15];
    const float* ln1_g = (const float*)d_in[16]; const float* ln1_b = (const float*)d_in[17];
    const float* w1 = (const float*)d_in[18];  const float* b1 = (const float*)d_in[19];
    const float* w2 = (const float*)d_in[20];  const float* b2 = (const float*)d_in[21];
    const float* ln2_g = (const float*)d_in[22]; const float* ln2_b = (const float*)d_in[23];
    const float* cls_w1 = (const float*)d_in[24]; const float* cls_b1 = (const float*)d_in[25];
    const float* cls_w2 = (const float*)d_in[26]; const float* cls_b2 = (const float*)d_in[27];
    const int*   rand_blocks = (const int*)d_in[28];

    char* wsp = (char*)d_ws;
    size_t off = 0;
    auto alloc = [&](size_t bytes) -> void* {
        void* p = wsp + off;
        off += (bytes + 255) & ~(size_t)255;
        return p;
    };
    float*          mask_add = (float*)alloc((size_t)S_LEN * 4);
    unsigned short* x_bf     = (unsigned short*)alloc((size_t)S_LEN * IN_DIM * 2);
    unsigned short* proj_wt  = (unsigned short*)alloc((size_t)D_MODEL * IN_DIM * 2);
    unsigned short* wq_t = (unsigned short*)alloc((size_t)N_LAYER * D_MODEL * D_MODEL * 2);
    unsigned short* wk_t = (unsigned short*)alloc((size_t)N_LAYER * D_MODEL * D_MODEL * 2);
    unsigned short* wv_t = (unsigned short*)alloc((size_t)N_LAYER * D_MODEL * D_MODEL * 2);
    unsigned short* wo_t = (unsigned short*)alloc((size_t)N_LAYER * D_MODEL * D_MODEL * 2);
    unsigned short* w1_t = (unsigned short*)alloc((size_t)N_LAYER * FF_DIM * D_MODEL * 2);
    unsigned short* w2_t = (unsigned short*)alloc((size_t)N_LAYER * D_MODEL * FF_DIM * 2);
    unsigned short* c1_t = (unsigned short*)alloc((size_t)H1_DIM * D_MODEL * 2);
    unsigned short* c2_t = (unsigned short*)alloc((size_t)TGT_DIM * H1_DIM * 2);
    int*            blist = (int*)alloc((size_t)N_HEAD * NB_BLK * BL_STRIDE * 4);
    float*          h     = (float*)alloc((size_t)S_LEN * D_MODEL * 4);
    unsigned short* h_bf  = (unsigned short*)alloc((size_t)S_LEN * D_MODEL * 2);
    float*          tmp   = (float*)alloc((size_t)S_LEN * D_MODEL * 4);
    unsigned short* q_bf  = (unsigned short*)alloc((size_t)S_LEN * D_MODEL * 2);
    unsigned short* k_bf  = (unsigned short*)alloc((size_t)S_LEN * D_MODEL * 2);
    unsigned short* v_bf  = (unsigned short*)alloc((size_t)S_LEN * D_MODEL * 2);
    unsigned short* ctx_bf = (unsigned short*)alloc((size_t)S_LEN * D_MODEL * 2);
    unsigned short* f_bf  = (unsigned short*)alloc((size_t)S_LEN * FF_DIM * 2);
    unsigned short* z_bf  = (unsigned short*)alloc((size_t)128 * H1_DIM * 2);

    auto gemm = [&](const unsigned short* Aa, const unsigned short* Bt, const float* bias,
                    float* Cf, unsigned short* Cb, int M, int N, int K, int act) {
        dim3 grid((N + 127) / 128, (M + 127) / 128);
        gemm_bf16<<<grid, 128, 0, stream>>>(Aa, Bt, bias, Cf, Cb, M, N, K, act);
    };
    auto wconv = [&](const float* W, unsigned short* Wt, int K, int N) {
        int n = K * N;
        wt_kernel<<<(n + 255) / 256, 256, 0, stream>>>(W, Wt, K, N);
    };

    // --- prep ---
    mask_kernel<<<(S_LEN + 255) / 256, 256, 0, stream>>>(am, mask_add, S_LEN);
    f32_to_bf16_kernel<<<((S_LEN * IN_DIM) + 255) / 256, 256, 0, stream>>>(
        x, x_bf, S_LEN * IN_DIM);
    wconv(proj_w, proj_wt, IN_DIM, D_MODEL);
    for (int i = 0; i < N_LAYER; ++i) {
        size_t wofs = (size_t)i * D_MODEL * D_MODEL;
        size_t fofs = (size_t)i * D_MODEL * FF_DIM;
        wconv(wq + wofs, wq_t + wofs, D_MODEL, D_MODEL);
        wconv(wk + wofs, wk_t + wofs, D_MODEL, D_MODEL);
        wconv(wv + wofs, wv_t + wofs, D_MODEL, D_MODEL);
        wconv(wo + wofs, wo_t + wofs, D_MODEL, D_MODEL);
        wconv(w1 + fofs, w1_t + fofs, D_MODEL, FF_DIM);
        wconv(w2 + fofs, w2_t + fofs, FF_DIM, D_MODEL);
    }
    wconv(cls_w1, c1_t, D_MODEL, H1_DIM);
    wconv(cls_w2, c2_t, H1_DIM, TGT_DIM);
    build_blist<<<(N_HEAD * NB_BLK + 255) / 256, 256, 0, stream>>>(rand_blocks, blist);

    // --- embedding ---
    gemm(x_bf, proj_wt, proj_b, tmp, nullptr, S_LEN, D_MODEL, IN_DIM, 0);
    embadd_kernel<<<((S_LEN * D_MODEL) + 255) / 256, 256, 0, stream>>>(
        tmp, pos_emb, type_emb, S_LEN * D_MODEL, D_MODEL);
    ln_kernel<<<S_LEN, 256, 0, stream>>>(tmp, nullptr, emb_g, emb_b, h, h_bf, D_MODEL);

    // --- transformer layers ---
    for (int i = 0; i < N_LAYER; ++i) {
        size_t wofs = (size_t)i * D_MODEL * D_MODEL;
        size_t fofs = (size_t)i * D_MODEL * FF_DIM;
        size_t dofs = (size_t)i * D_MODEL;
        size_t ffofs = (size_t)i * FF_DIM;

        gemm(h_bf, wq_t + wofs, bq + dofs, nullptr, q_bf, S_LEN, D_MODEL, D_MODEL, 0);
        gemm(h_bf, wk_t + wofs, bk + dofs, nullptr, k_bf, S_LEN, D_MODEL, D_MODEL, 0);
        gemm(h_bf, wv_t + wofs, bv + dofs, nullptr, v_bf, S_LEN, D_MODEL, D_MODEL, 0);

        dim3 ag(N_HEAD, NB_BLK);
        bigbird_attn<<<ag, 128, 0, stream>>>(q_bf, k_bf, v_bf, mask_add, blist, ctx_bf);

        gemm(ctx_bf, wo_t + wofs, bo + dofs, tmp, nullptr, S_LEN, D_MODEL, D_MODEL, 0);
        ln_kernel<<<S_LEN, 256, 0, stream>>>(tmp, h, ln1_g + dofs, ln1_b + dofs,
                                             h, h_bf, D_MODEL);

        gemm(h_bf, w1_t + fofs, b1 + ffofs, nullptr, f_bf, S_LEN, FF_DIM, D_MODEL, 1);
        gemm(f_bf, w2_t + fofs, b2 + dofs, tmp, nullptr, S_LEN, D_MODEL, FF_DIM, 0);
        ln_kernel<<<S_LEN, 256, 0, stream>>>(tmp, h, ln2_g + dofs, ln2_b + dofs,
                                             h, h_bf, D_MODEL);
    }

    // --- classifier head on CLS token (row 0) ---
    gemm(h_bf, c1_t, cls_b1, nullptr, z_bf, 1, H1_DIM, D_MODEL, 2);
    gemm(z_bf, c2_t, cls_b2, (float*)d_out, nullptr, 1, TGT_DIM, H1_DIM, 0);
}
